// HungarianMatcher_31791347925741
// MI455X (gfx1250) — matrix operand order repaired and checked by reading, not compiled, not run on hardware
//
#include <hip/hip_runtime.h>
#include <hip/hip_bf16.h>
#include <math.h>

// Problem constants (from reference)
#define BATCH 16
#define NPRED 900
#define NCLS  91
#define NTGT  100
// Padded GEMM dims: A = softmax probs (NP x KP), B = -onehot (KP x MP)
#define NP 912     // 57 tiles of 16
#define KP 96      // 3 k-steps of 32
#define MP 112     // 7 tiles of 16
#define LAMBDA_L1  5.0f
#define LAMBDA_IOU 2.0f
#define LAMBDA_CLS 1.0f

typedef __attribute__((ext_vector_type(16))) _Float16 v16h;
typedef __attribute__((ext_vector_type(8)))  float    v8f;

// ---------------------------------------------------------------------------
// Kernel 1: softmax over K=91 classes, one wave32 per (b, n) row.
// Writes f16 probs into P16[b][n][k], zero-padded to KP=96 and NP=912 rows.
// ---------------------------------------------------------------------------
__global__ __launch_bounds__(256)
void softmax_rows_kernel(const float* __restrict__ cls_pred,
                         _Float16* __restrict__ P16) {
  const int row  = blockIdx.x * 8 + (threadIdx.x >> 5);   // 0 .. BATCH*NP-1
  const int lane = threadIdx.x & 31;
  if (row >= BATCH * NP) return;
  const int b = row / NP;
  const int n = row % NP;
  _Float16* dst = P16 + ((size_t)b * NP + n) * KP;

  if (n >= NPRED) {  // pad rows -> zero probs
    dst[lane]      = (_Float16)0.0f;
    dst[lane + 32] = (_Float16)0.0f;
    dst[lane + 64] = (_Float16)0.0f;
    return;
  }

  const float* src = cls_pred + ((size_t)b * NPRED + n) * NCLS;
  const int k0 = lane, k1 = lane + 32, k2 = lane + 64;
  const float x0 = (k0 < NCLS) ? src[k0] : -1e30f;
  const float x1 = (k1 < NCLS) ? src[k1] : -1e30f;
  const float x2 = (k2 < NCLS) ? src[k2] : -1e30f;

  float mx = fmaxf(fmaxf(x0, x1), x2);
  #pragma unroll
  for (int off = 16; off > 0; off >>= 1)
    mx = fmaxf(mx, __shfl_xor(mx, off, 32));

  const float e0 = (k0 < NCLS) ? __expf(x0 - mx) : 0.0f;
  const float e1 = (k1 < NCLS) ? __expf(x1 - mx) : 0.0f;
  const float e2 = (k2 < NCLS) ? __expf(x2 - mx) : 0.0f;
  float s = e0 + e1 + e2;
  #pragma unroll
  for (int off = 16; off > 0; off >>= 1)
    s += __shfl_xor(s, off, 32);
  const float inv = 1.0f / s;

  dst[k0] = (_Float16)(e0 * inv);
  dst[k1] = (_Float16)(e1 * inv);
  dst[k2] = (_Float16)(e2 * inv);
}

// ---------------------------------------------------------------------------
// Kernel 2a/2b: build the m-major negative one-hot matrix OHT[b][m][k] (f16):
// OHT[b][m][cls_gt[b][m]] = -1, else 0.  (B operand of the WMMA GEMM.)
// ---------------------------------------------------------------------------
__global__ __launch_bounds__(256)
void zero_oht_kernel(unsigned int* __restrict__ oht_u32) {
  const int i = blockIdx.x * blockDim.x + threadIdx.x;
  if (i < BATCH * MP * KP / 2) oht_u32[i] = 0u;
}

__global__ __launch_bounds__(256)
void set_oht_kernel(const int* __restrict__ cls_gt, _Float16* __restrict__ OHT) {
  const int i = blockIdx.x * blockDim.x + threadIdx.x;
  if (i >= BATCH * NTGT) return;
  const int b = i / NTGT;
  const int m = i % NTGT;
  const int c = cls_gt[b * NTGT + m];
  OHT[((size_t)b * MP + m) * KP + c] = (_Float16)(-1.0f);
}

// ---------------------------------------------------------------------------
// Kernel 3: fused cost-tile kernel. One wave per 16x16 output tile.
//   acc(16x16) = P16(16xKP) x OHT^T(KPx16) via v_wmma_f32_16x16x32_f16
//   (acc == cost_class because OHT holds -1)
// then per output element add 5*L1 - 2*IoU and store f32 cost.
// ---------------------------------------------------------------------------
__global__ __launch_bounds__(32)
void cost_tile_kernel(const _Float16* __restrict__ P16,
                      const _Float16* __restrict__ OHT,
                      const float* __restrict__ bb_pred,
                      const float* __restrict__ bb_gt,
                      float* __restrict__ C) {
  const int n0   = blockIdx.x * 16;
  const int m0   = blockIdx.y * 16;
  const int b    = blockIdx.z;
  const int lane = threadIdx.x;
  const int hi   = (lane & 16) ? 1 : 0;

  // A fragment source: row = n0 + (lane&15); this code's contiguous k order (same as B):
  //   K = kbase + h, kbase = 16 for lanes 16..31.
  const int arow = n0 + (lane & 15);
  const _Float16* Arow = P16 + ((size_t)b * NP + arow) * KP + (hi ? 16 : 0);
  // B fragment source: col = m0 + (lane&15); 16 consecutive K values,
  // lanes 16..31 carry K+16. OHT is m-major so this is one 32B vector load.
  const int bcol = m0 + (lane & 15);
  const _Float16* Brow = OHT + ((size_t)b * MP + bcol) * KP + (hi ? 16 : 0);

  v8f acc = {};
  #pragma unroll
  for (int ks = 0; ks < 3; ++ks) {
    const int k0 = ks * 32;
    v16h a;
    #pragma unroll
    for (int j = 0; j < 8; ++j) {
      const int kp = k0 + 2 * j;
      a[2 * j]     = Arow[kp];
      a[2 * j + 1] = Arow[kp + 1];
    }
    const v16h bf = *(const v16h*)(Brow + k0);
    acc = __builtin_amdgcn_wmma_f32_16x16x32_f16(
        /*neg_a=*/false, a, /*neg_b=*/false, bf,
        /*c_mod=*/(short)0, acc, /*reuse_a=*/false, /*reuse_b=*/false);
  }

  // Geometry terms + store. C/D layout: VGPR r -> row n0+r (+8 for lanes>=16),
  // col = m0 + (lane&15).
  const int m = m0 + (lane & 15);
  float g0 = 0.f, g1 = 0.f, g2 = 0.f, g3 = 0.f, garea = 0.f;
  const bool mval = (m < NTGT);
  if (mval) {
    const float* g = bb_gt + ((size_t)b * NTGT + m) * 4;
    g0 = g[0]; g1 = g[1]; g2 = g[2]; g3 = g[3];
    garea = (g2 - g0) * (g3 - g1);
  }
  #pragma unroll
  for (int r = 0; r < 8; ++r) {
    const int n = n0 + r + (hi ? 8 : 0);
    float cost = 1e30f;  // padding entries: never read by solver
    if (n < NPRED && mval) {
      const float* p = bb_pred + ((size_t)b * NPRED + n) * 4;
      const float p0 = p[0], p1 = p[1], p2 = p[2], p3 = p[3];
      const float l1 = fabsf(p0 - g0) + fabsf(p1 - g1) +
                       fabsf(p2 - g2) + fabsf(p3 - g3);
      const float parea = (p2 - p0) * (p3 - p1);
      const float iw = fminf(p2, g2) - fmaxf(p0, g0);
      const float ih = fminf(p3, g3) - fmaxf(p1, g1);
      const float inter = fmaxf(iw, 0.f) * fmaxf(ih, 0.f);
      const float iou = inter / (parea + garea - inter + 1e-6f);
      cost = LAMBDA_CLS * acc[r] + LAMBDA_L1 * l1 - LAMBDA_IOU * iou;
    }
    C[((size_t)b * NP + n) * MP + m] = cost;
  }
}

// ---------------------------------------------------------------------------
// Kernel 4: Jonker-Volgenant (same algorithm as reference _solve) on the
// transposed 100x900 problem. One workgroup (256 threads) per batch; all
// solver state in LDS (~18 KB << 320 KB/WGP). Block-wide lexicographic
// (value, index) argmin reproduces np.argmin first-min tie-breaking.
// ---------------------------------------------------------------------------
#define HTHREADS 256
__global__ __launch_bounds__(HTHREADS)
void hungarian_kernel(const float* __restrict__ C, long long* __restrict__ out) {
  const int b   = blockIdx.x;
  const int tid = threadIdx.x;
  const float* Cb = C + (size_t)b * NP * MP;   // Cb[n*MP + m]

  __shared__ float sv[NPRED + 1];     // v  (col potentials), col = pred index
  __shared__ float sminv[NPRED + 1];
  __shared__ float su[NTGT + 1];      // u  (row potentials), row = target
  __shared__ int   sp[NPRED + 1];     // p  (col -> row)
  __shared__ int   sway[NPRED + 1];
  __shared__ unsigned char sused[NPRED + 1];
  __shared__ float redv[HTHREADS];
  __shared__ int   redi[HTHREADS];
  __shared__ int   sh_j0, sh_i0;
  __shared__ float sh_delta;

  for (int j = tid; j <= NPRED; j += HTHREADS) { sv[j] = 0.f; sp[j] = 0; sway[j] = 0; }
  for (int i = tid; i <= NTGT;  i += HTHREADS) su[i] = 0.f;
  __syncthreads();

  for (int i = 1; i <= NTGT; ++i) {
    for (int j = tid; j <= NPRED; j += HTHREADS) { sminv[j] = 1e30f; sused[j] = 0; }
    if (tid == 0) { sp[0] = i; sh_j0 = 0; }
    __syncthreads();

    for (;;) {
      if (tid == 0) { sused[sh_j0] = 1; sh_i0 = sp[sh_j0]; }
      __syncthreads();
      const int   i0 = sh_i0;
      const int   j0 = sh_j0;
      const float ui = su[i0];

      float best = 1e30f; int bestj = NPRED + 2;
      for (int j = tid + 1; j <= NPRED; j += HTHREADS) {
        if (!sused[j]) {
          const float cur = Cb[(size_t)(j - 1) * MP + (i0 - 1)] - ui - sv[j];
          if (cur < sminv[j]) { sminv[j] = cur; sway[j] = j0; }
          const float mv = sminv[j];
          if (mv < best) { best = mv; bestj = j; }
        }
      }
      redv[tid] = best; redi[tid] = bestj;
      __syncthreads();
      #pragma unroll
      for (int s = HTHREADS / 2; s > 0; s >>= 1) {
        if (tid < s) {
          const float ov = redv[tid + s]; const int oi = redi[tid + s];
          if (ov < redv[tid] || (ov == redv[tid] && oi < redi[tid])) {
            redv[tid] = ov; redi[tid] = oi;
          }
        }
        __syncthreads();
      }
      if (tid == 0) { sh_delta = redv[0]; sh_j0 = redi[0]; }
      __syncthreads();

      const float delta = sh_delta;
      for (int j = tid; j <= NPRED; j += HTHREADS) {
        if (sused[j]) { su[sp[j]] += delta; sv[j] -= delta; }
        else          { sminv[j] -= delta; }
      }
      __syncthreads();
      if (sp[sh_j0] == 0) break;
    }

    if (tid == 0) {  // augment along the alternating path
      int j0 = sh_j0;
      while (j0 != 0) { const int j1 = sway[j0]; sp[j0] = sp[j1]; j0 = j1; }
    }
    __syncthreads();
  }

  if (tid == 0) {
    long long* preds = out;                              // (BATCH, NTGT)
    long long* tgts  = out + (size_t)BATCH * NTGT;       // (BATCH, NTGT)
    int cnt = 0;
    for (int j = 1; j <= NPRED && cnt < NTGT; ++j) {
      if (sp[j] != 0) {   // naturally sorted by pred index
        preds[b * NTGT + cnt] = (long long)(j - 1);
        tgts[b * NTGT + cnt]  = (long long)(sp[j] - 1);
        ++cnt;
      }
    }
  }
}

// ---------------------------------------------------------------------------
extern "C" void kernel_launch(void* const* d_in, const int* in_sizes, int n_in,
                              void* d_out, int out_size, void* d_ws, size_t ws_size,
                              hipStream_t stream) {
  const float* cls_pred = (const float*)d_in[0];
  const float* bb_pred  = (const float*)d_in[1];
  const int*   cls_gt   = (const int*)d_in[2];
  const float* bb_gt    = (const float*)d_in[3];

  // Workspace carve-up
  char* ws = (char*)d_ws;
  _Float16* P16 = (_Float16*)ws;                               // B*NP*KP f16
  ws += (size_t)BATCH * NP * KP * sizeof(_Float16);
  _Float16* OHT = (_Float16*)ws;                               // B*MP*KP f16
  ws += (size_t)BATCH * MP * KP * sizeof(_Float16);
  float* C = (float*)ws;                                       // B*NP*MP f32

  // 1) softmax -> f16 padded probs
  {
    const int rows = BATCH * NP;
    softmax_rows_kernel<<<rows / 8, 256, 0, stream>>>(cls_pred, P16);
  }
  // 2) negative one-hot (m-major)
  {
    const int dwords = BATCH * MP * KP / 2;
    zero_oht_kernel<<<(dwords + 255) / 256, 256, 0, stream>>>((unsigned int*)OHT);
    set_oht_kernel<<<(BATCH * NTGT + 255) / 256, 256, 0, stream>>>(cls_gt, OHT);
  }
  // 3) fused WMMA class-cost GEMM + L1/IoU
  {
    dim3 grid(NP / 16, MP / 16, BATCH);
    cost_tile_kernel<<<grid, 32, 0, stream>>>(P16, OHT, bb_pred, bb_gt, C);
  }
  // 4) per-batch LAPJV solve + int64 output
  hungarian_kernel<<<BATCH, HTHREADS, 0, stream>>>(C, (long long*)d_out);
}
